// Quantizer_12575664243240
// MI455X (gfx1250) — matrix operand order repaired and checked by reading, not compiled, not run on hardware
//
#include <hip/hip_runtime.h>

// ---------------------------------------------------------------------------
// VQ quantizer for MI455X (gfx1250, wave32, WMMA + async-to-LDS).
// score(token, code) = ||c||^2 - 2*dot(ze, c)   (||ze||^2 constant per token)
// dot via split-f16 WMMA: x = hi + lo (f16), dot ~= ah*bh + ah*bl + al*bh
// (three independent f32-accumulating WMMA chains -> ~fp32-accurate argmin).
// hi/lo planes interleaved per row (stride 1024 halfs); codebook tiles live
// at LDS offset 0 and are double-buffered + filled with
// global_load_async_to_lds_b128 (ASYNCcnt); kt-loop unrolled x2 and base
// indices laundered so every fragment load is ds_load_b128 [base + imm].
// ---------------------------------------------------------------------------

typedef _Float16 h16 __attribute__((ext_vector_type(16)));
typedef _Float16 h8  __attribute__((ext_vector_type(8)));
typedef _Float16 h4  __attribute__((ext_vector_type(4)));
typedef float    f8  __attribute__((ext_vector_type(8)));

#define DIM        512
#define ROWS       1024                  // interleaved row stride: hi|lo halfs
#define KCODES     8192
#define NTOK       (16 * 2048)
#define TOK_PER_WG 64
#define BLOCK      128                   // 4 waves of 32
#define KT         16                    // codes per tile
#define NKT        (KCODES / KT)
#define TILE_H     (KT * ROWS)           // halfs per codebook tile (32 KB)

// Async 16-byte copy global -> LDS, tracked with ASYNCcnt.
// Generic LDS pointers are {shared_aperture, lds_offset}; the low 32 bits are
// the LDS-relative byte address the instruction's VDST expects.
__device__ __forceinline__ void async_b128(const void* gsrc, void* ldst) {
  unsigned loff = (unsigned)(unsigned long long)(uintptr_t)ldst;
  unsigned long long gaddr = (unsigned long long)(uintptr_t)gsrc;
  asm volatile("global_load_async_to_lds_b128 %0, %1, off"
               :: "v"(loff), "v"(gaddr)
               : "memory");
}

__device__ __forceinline__ void wait_async0() {
#if __has_builtin(__builtin_amdgcn_s_wait_asynccnt)
  __builtin_amdgcn_s_wait_asynccnt(0);
#else
  asm volatile("s_wait_asynccnt 0" ::: "memory");
#endif
}

// Force a value into a VGPR the compiler cannot re-split, so region-base
// constants stay folded into the base register instead of (illegally) into
// 16-bit ds_load immediates (which forces per-access v_add rematerialization).
__device__ __forceinline__ int opaque(int v) {
  asm("" : "+v"(v));
  return v;
}

// One 32 KB codebook tile, streamed as 2048 contiguous async 16 B chunks.
__device__ __forceinline__ void stage_tile(const _Float16* __restrict__ gT,
                                           _Float16* lT) {
#pragma unroll
  for (int i = 0; i < TILE_H / 8 / BLOCK; ++i) {
    const int c = threadIdx.x + i * BLOCK;
    async_b128(gT + (size_t)c * 8, lT + (size_t)c * 8);
  }
}

// 16x16 score tile vs 16 codes: 16 d-chunks x 3 split-term WMMAs, then fold
// into the per-lane running argmin. bBase carries a compile-time buffer
// offset so all LDS loads are [base + imm].
__device__ __forceinline__ void compute_tile(const _Float16* aBase,
                                             const _Float16* bBase, float cn,
                                             int kbase, int n, float best[8],
                                             int bidx[8]) {
  f8 acc_hh = {};
  f8 acc_hl = {};
  f8 acc_lh = {};
#pragma unroll 4
  for (int d = 0; d < DIM; d += 32) {
    h8 a0 = *(const h8*)(aBase + d);
    h8 a1 = *(const h8*)(aBase + d + 16);
    h16 ah = __builtin_shufflevector(a0, a1, 0, 1, 2, 3, 4, 5, 6, 7, 8, 9, 10,
                                     11, 12, 13, 14, 15);
    h8 c0 = *(const h8*)(aBase + DIM + d);
    h8 c1 = *(const h8*)(aBase + DIM + d + 16);
    h16 al = __builtin_shufflevector(c0, c1, 0, 1, 2, 3, 4, 5, 6, 7, 8, 9, 10,
                                     11, 12, 13, 14, 15);
    h8 b0 = *(const h8*)(bBase + d);
    h8 b1 = *(const h8*)(bBase + d + 16);
    h16 bh = __builtin_shufflevector(b0, b1, 0, 1, 2, 3, 4, 5, 6, 7, 8, 9, 10,
                                     11, 12, 13, 14, 15);
    h8 e0 = *(const h8*)(bBase + DIM + d);
    h8 e1 = *(const h8*)(bBase + DIM + d + 16);
    h16 bl = __builtin_shufflevector(e0, e1, 0, 1, 2, 3, 4, 5, 6, 7, 8, 9, 10,
                                     11, 12, 13, 14, 15);
    acc_hh = __builtin_amdgcn_wmma_f32_16x16x32_f16(false, ah, false, bh,
                                                    (short)0, acc_hh, false,
                                                    false);
    acc_hl = __builtin_amdgcn_wmma_f32_16x16x32_f16(false, ah, false, bl,
                                                    (short)0, acc_hl, false,
                                                    false);
    acc_lh = __builtin_amdgcn_wmma_f32_16x16x32_f16(false, al, false, bh,
                                                    (short)0, acc_lh, false,
                                                    false);
  }
  // C/D layout: VGPR v -> token (v + 8*hh), col n. Running min over k.
#pragma unroll
  for (int v = 0; v < 8; ++v) {
    float s = cn - 2.0f * (acc_hh[v] + acc_hl[v] + acc_lh[v]);
    if (s < best[v]) { best[v] = s; bidx[v] = kbase + n; }
  }
}

// ---------------------------------------------------------------------------
// Prep: split codebook rows into interleaved f16 [hi 512 | lo 512] rows and
// compute ||c_k||^2 (fp32). One wave per row; 4 rows per 128-thread block.
// ---------------------------------------------------------------------------
__global__ __launch_bounds__(BLOCK) void vq_prep(
    const float* __restrict__ cb, float* __restrict__ cnorm,
    _Float16* __restrict__ cbSplit) {
  const int lane = threadIdx.x & 31;
  const int wave = threadIdx.x >> 5;
  const int row  = blockIdx.x * 4 + wave;

  const float4* src = (const float4*)(cb + (size_t)row * DIM);
  h4* dstHi = (h4*)(cbSplit + (size_t)row * ROWS);
  h4* dstLo = (h4*)(cbSplit + (size_t)row * ROWS + DIM);

  float sum = 0.0f;
#pragma unroll
  for (int c = 0; c < 4; ++c) {
    const int chunk = lane + c * 32;          // 128 float4 chunks per row
    float4 x = src[chunk];
    sum += x.x * x.x + x.y * x.y + x.z * x.z + x.w * x.w;
    h4 hi = {(_Float16)x.x, (_Float16)x.y, (_Float16)x.z, (_Float16)x.w};
    h4 lo = {(_Float16)(x.x - (float)hi[0]), (_Float16)(x.y - (float)hi[1]),
             (_Float16)(x.z - (float)hi[2]), (_Float16)(x.w - (float)hi[3])};
    dstHi[chunk] = hi;
    dstLo[chunk] = lo;
  }
#pragma unroll
  for (int off = 16; off > 0; off >>= 1) sum += __shfl_xor(sum, off, 32);
  if (lane == 0) cnorm[row] = sum;
}

// ---------------------------------------------------------------------------
// Main: 4 waves x 16 tokens = 64 tokens per WG.
// LDS: [ 2 x 32 KB codebook tiles | 128 KB ze tile ] = 192 KB.
// ---------------------------------------------------------------------------
__global__ __launch_bounds__(BLOCK) void vq_main(
    const float* __restrict__ ze, const float* __restrict__ cbF32,
    const float* __restrict__ cnorm, const _Float16* __restrict__ cbSplitG,
    float* __restrict__ out) {
  extern __shared__ _Float16 smem[];
  _Float16* cbTiles = smem;                        // 2 x (16 x 1024) halfs
  _Float16* zeT     = smem + 2 * TILE_H;           // 64 rows x 1024 halfs

  const int lane = threadIdx.x & 31;
  const int wave = threadIdx.x >> 5;
  const int n    = lane & 15;   // token row (A/C) and code col (B/C) index
  const int hh   = lane >> 4;   // which 16-lane half

  // --- stage ze tile once, f32 -> interleaved f16 hi|lo rows --------------
  const float4* zeG =
      (const float4*)(ze + (size_t)blockIdx.x * TOK_PER_WG * DIM);
  for (int i = threadIdx.x; i < TOK_PER_WG * DIM / 4; i += BLOCK) {
    const int row  = i >> 7;          // 128 float4 chunks per 512-float row
    const int col4 = i & 127;
    float4 x = zeG[i];
    h4 hi = {(_Float16)x.x, (_Float16)x.y, (_Float16)x.z, (_Float16)x.w};
    h4 lo = {(_Float16)(x.x - (float)hi[0]), (_Float16)(x.y - (float)hi[1]),
             (_Float16)(x.z - (float)hi[2]), (_Float16)(x.w - (float)hi[3])};
    ((h4*)(zeT + row * ROWS))[col4]       = hi;
    ((h4*)(zeT + row * ROWS + DIM))[col4] = lo;
  }

  // --- prologue: async-fill tile 0 into buffer 0 --------------------------
  stage_tile(cbSplitG, cbTiles);

  float best[8];
  int   bidx[8];
#pragma unroll
  for (int v = 0; v < 8; ++v) { best[v] = 3.4e38f; bidx[v] = 0; }

  // A (16x32 f16): lane m=n, K runs [d+8*hh .. +7], [d+16+8*hh .. +7];
  // lo plane at +DIM halfs off the same base register. The 2*TILE_H region
  // base (64 KB) is laundered into the base VGPR so it cannot leak into the
  // 16-bit ds_load immediates.
  const _Float16* aBase =
      smem + opaque(2 * TILE_H + (wave * 16 + n) * ROWS + 8 * hh);
  // B (32x16 f16): lane col=n, K runs [d+8*hh .. +7], [d+16+8*hh .. +7]; lo at +DIM.
  // Both tile buffers reachable from this one register via immediates
  // (buffer 1 = +TILE_H halfs = +32 KB).
  const _Float16* bBase = smem + opaque(n * ROWS + 8 * hh);

  for (int kt = 0; kt < NKT; kt += 2) {
    // ---- even tile: compute buffer 0, prefetch kt+1 into buffer 1 --------
    wait_async0();          // own tile-kt async loads landed
    __syncthreads();        // everyone's landed; prior readers of buf1 done
    if (kt + 1 < NKT)
      stage_tile(cbSplitG + (size_t)(kt + 1) * TILE_H, cbTiles + TILE_H);
    compute_tile(aBase, bBase, cnorm[kt * KT + n], kt * KT, n, best, bidx);

    // ---- odd tile: compute buffer 1, prefetch kt+2 into buffer 0 ---------
    wait_async0();
    __syncthreads();
    if (kt + 2 < NKT)
      stage_tile(cbSplitG + (size_t)(kt + 2) * TILE_H, cbTiles);
    compute_tile(aBase, bBase + TILE_H, cnorm[(kt + 1) * KT + n],
                 (kt + 1) * KT, n, best, bidx);
  }

  // --- min+argmin across the 16 lanes of each half ------------------------
#pragma unroll
  for (int v = 0; v < 8; ++v) {
#pragma unroll
    for (int off = 1; off < 16; off <<= 1) {
      float ob = __shfl_xor(best[v], off, 32);
      int   oi = __shfl_xor(bidx[v], off, 32);
      if (ob < best[v] || (ob == best[v] && oi < bidx[v])) {
        best[v] = ob;
        bidx[v] = oi;
      }
    }
  }

  // --- gather: zq[token] = codebook[argmin]; wave copies 2 KB per token ---
  const size_t tokBase = (size_t)blockIdx.x * TOK_PER_WG + (size_t)wave * 16;
#pragma unroll
  for (int t = 0; t < 16; ++t) {
    const int v   = t & 7;
    const int idx = __shfl(bidx[v], (t < 8) ? 0 : 16, 32);
    const float4* src = (const float4*)(cbF32 + (size_t)idx * DIM);
    float4*       dst = (float4*)(out + (tokBase + t) * DIM);
#pragma unroll
    for (int c = 0; c < 4; ++c) dst[lane + c * 32] = src[lane + c * 32];
  }
}

// ---------------------------------------------------------------------------
// d_ws layout: [ cnorm f32[8192] | cbSplit f16[8192][1024] (hi|lo rows) ]
//              = 32 KB + 16 MB ~= 16.03 MB required.
// ---------------------------------------------------------------------------
extern "C" void kernel_launch(void* const* d_in, const int* in_sizes, int n_in,
                              void* d_out, int out_size, void* d_ws,
                              size_t ws_size, hipStream_t stream) {
  (void)in_sizes; (void)n_in; (void)out_size; (void)ws_size;
  const float* ze = (const float*)d_in[0];
  const float* cb = (const float*)d_in[1];
  float* out = (float*)d_out;

  float*    cnorm   = (float*)d_ws;
  _Float16* cbSplit = (_Float16*)((char*)d_ws + (size_t)KCODES * sizeof(float));

  vq_prep<<<KCODES / 4, BLOCK, 0, stream>>>(cb, cnorm, cbSplit);

  const size_t lds = (size_t)(TOK_PER_WG * ROWS + 2 * TILE_H) *
                     sizeof(_Float16);  // 192 KB
  vq_main<<<NTOK / TOK_PER_WG, BLOCK, lds, stream>>>(ze, cb, cnorm, cbSplit,
                                                     out);
}
